// MambaLayer_62508954026213
// MI455X (gfx1250) — compile-verified, matrix operand order checked by reading, not run on hardware
//
#include <hip/hip_runtime.h>
#include <hip/hip_bf16.h>

// ---------------- problem constants (fixed by the reference) ----------------
#define BATCH    2
#define LSEQ     2048
#define DMODEL   1024
#define DSTATE   16
#define DCONV    4
#define DINNER   2048          // EXPAND * DMODEL
#define DTRANK   64
#define ROWS     (BATCH * LSEQ)   // 4096 token rows

// ---------------- WMMA tile config ----------------
#define TILE_M 128
#define TILE_N 128
#define TILE_K 32
#define LDS_STRIDE (TILE_K + 8)   // 40 halves = 80B rows (TDM pads 16B/row)

typedef __attribute__((ext_vector_type(16))) _Float16 v16h;
typedef __attribute__((ext_vector_type(8)))  _Float16 v8h;
typedef __attribute__((ext_vector_type(8)))  float    v8f;
typedef __attribute__((ext_vector_type(4)))  unsigned int v4u;
typedef __attribute__((ext_vector_type(8)))  int      v8i;
typedef __attribute__((ext_vector_type(4)))  int      v4i;

union V16 { v16h v; v8h p[2]; };

__device__ inline v8f v8f_zero() {
  v8f r;
#pragma unroll
  for (int i = 0; i < 8; ++i) r[i] = 0.0f;
  return r;
}

// =====================================================================
// TDM: issue one 2D tile load (f16 elements) global -> LDS.
// Builds the D# descriptor per CDNA5 ISA 8.3/8.4:
//   tile_d0 elements per row (64B here), tile_d1 rows, row stride in
//   elements; LDS destination padded +4 DWORDs after every 16 DWORDs
//   (row stride 80B) to match the Al/Bl layout below.
// 6-arg builtin: (v4u group0, v8i group1, v4i group2, v4i group3,
//                 v8i extra, int cpol)
// =====================================================================
__device__ inline void tdm_load_tile_f16(unsigned lds_off, const void* gptr,
                                         unsigned tensor_d0, unsigned tensor_d1,
                                         unsigned tile_d0, unsigned tile_d1,
                                         unsigned long long stride_elems)
{
  unsigned long long ga = (unsigned long long)gptr;
  v4u g0;
  g0[0] = 1u;                                       // count=1 (valid user desc)
  g0[1] = lds_off;                                  // LDS byte address
  g0[2] = (unsigned)(ga & 0xFFFFFFFFu);             // global_addr[31:0]
  g0[3] = (unsigned)((ga >> 32) & 0x01FFFFFFu)      // global_addr[56:32]
        | (2u << 30);                               // type=2 ("image")
  v8i g1;
  unsigned w0 = (1u << 16)    // data_size = 1 -> 2 bytes
              | (1u << 20)    // pad_enable
              | (3u << 22)    // pad_interval: 16 DWORDs (64B) between pads
              | (3u << 25);   // pad_amount:   4 DWORDs (16B)
  g1[0] = (int)w0;
  g1[1] = (int)((tensor_d0 & 0xFFFFu) << 16);                       // dim0[15:0]
  g1[2] = (int)(((tensor_d0 >> 16) & 0xFFFFu) |
                ((tensor_d1 & 0xFFFFu) << 16));                     // dim0[31:16] | dim1[15:0]
  g1[3] = (int)(((tensor_d1 >> 16) & 0xFFFFu) |
                ((tile_d0 & 0xFFFFu) << 16));                       // dim1[31:16] | tile_dim0
  g1[4] = (int)(tile_d1 & 0xFFFFu);                                 // tile_dim1 (tile_dim2=0)
  g1[5] = (int)(unsigned)(stride_elems & 0xFFFFFFFFull);            // dim0_stride[31:0]
  g1[6] = (int)(unsigned)((stride_elems >> 32) & 0xFFFFull);        // dim0_stride[47:32]
  g1[7] = 0;
  v4i z4 = {0, 0, 0, 0};
  v8i z8 = {0, 0, 0, 0, 0, 0, 0, 0};
  __builtin_amdgcn_tensor_load_to_lds(g0, g1, z4, z4, z8, 0);
}

// =====================================================================
// Generic f16 WMMA GEMM:  C[M,N] = A[M,K] * W[N,K]^T   (+ epilogue)
//   mode 0: C = acc
//   mode 1: C = acc + res               (residual add, res is [M,N] f32)
//   mode 2: C = softplus(acc + bias[n]) (dt projection)
// Block tile 128x128, 8 waves, 32x64 per wave (2x4 WMMA tiles).
// A/B tiles staged to LDS by the Tensor Data Mover (waves 0/1) with
// DOUBLE BUFFERING: TDM for tile k+1 overlaps WMMA on tile k; the
// s_wait_tensorcnt + barrier sit only at the iteration boundary.
// M % 128 == 0, K % 32 == 0 guaranteed by caller; N guarded.
// =====================================================================
__global__ __launch_bounds__(256) void wmma_gemm_kernel(
    const _Float16* __restrict__ A, const _Float16* __restrict__ Bw,
    float* __restrict__ C, int M, int N, int K, int mode,
    const float* __restrict__ bias, const float* __restrict__ res)
{
  __shared__ __align__(16) _Float16 Al[2][TILE_M][LDS_STRIDE];
  __shared__ __align__(16) _Float16 Bl[2][TILE_N][LDS_STRIDE];

  const int t     = threadIdx.x;
  const int wave  = t >> 5;
  const int lane  = t & 31;
  const int mw    = wave >> 1;        // 0..3  -> 32-row slab
  const int nw    = wave & 1;         // 0..1  -> 64-col slab
  const int lhalf = lane >> 4;        // 0/1 half-wave
  const int lrow  = lane & 15;        // row/col within 16
  const int k0    = lhalf * 8;        // K sub-group base per CDNA5 layout

  const int blockM = blockIdx.y * TILE_M;
  const int blockN = blockIdx.x * TILE_N;

  v8f acc[2][4];
#pragma unroll
  for (int i = 0; i < 2; ++i)
#pragma unroll
    for (int j = 0; j < 4; ++j) acc[i][j] = v8f_zero();

  // -------- prologue: stage K-tile 0 into buffer 0 --------
  if (wave == 0) {
    tdm_load_tile_f16((unsigned)(size_t)&Al[0][0][0],
                      &A[(size_t)blockM * K],
                      (unsigned)K, (unsigned)(M - blockM),
                      TILE_K, TILE_M, (unsigned long long)K);
  } else if (wave == 1) {
    tdm_load_tile_f16((unsigned)(size_t)&Bl[0][0][0],
                      &Bw[(size_t)blockN * K],
                      (unsigned)K, (unsigned)(N - blockN),
                      TILE_K, TILE_N, (unsigned long long)K);
  }
  __builtin_amdgcn_s_wait_tensorcnt(0);
  __syncthreads();

  int buf = 0;
  for (int kt = 0; kt < K; kt += TILE_K, buf ^= 1) {
    // -------- issue async DMA for the NEXT K-tile into the other buffer --
    const int nkt = kt + TILE_K;
    if (nkt < K) {
      if (wave == 0) {
        tdm_load_tile_f16((unsigned)(size_t)&Al[buf ^ 1][0][0],
                          &A[(size_t)blockM * K + nkt],
                          (unsigned)(K - nkt), (unsigned)(M - blockM),
                          TILE_K, TILE_M, (unsigned long long)K);
      } else if (wave == 1) {
        tdm_load_tile_f16((unsigned)(size_t)&Bl[buf ^ 1][0][0],
                          &Bw[(size_t)blockN * K + nkt],
                          (unsigned)(K - nkt), (unsigned)(N - blockN),
                          TILE_K, TILE_N, (unsigned long long)K);
      }
    }

    // -------- 2x4 WMMA tiles per wave from the CURRENT buffer --------
#pragma unroll
    for (int i = 0; i < 2; ++i) {
      V16 a;
      const _Float16* ap = &Al[buf][mw * 32 + i * 16 + lrow][0];
      a.p[0] = *(const v8h*)&ap[k0];
      a.p[1] = *(const v8h*)&ap[k0 + 16];
#pragma unroll
      for (int j = 0; j < 4; ++j) {
        V16 b;
        const _Float16* bp = &Bl[buf][nw * 64 + j * 16 + lrow][0];
        b.p[0] = *(const v8h*)&bp[k0];
        b.p[1] = *(const v8h*)&bp[k0 + 16];
        acc[i][j] = __builtin_amdgcn_wmma_f32_16x16x32_f16(
            /*neg_a=*/false, a.v, /*neg_b=*/false, b.v,
            /*c_mod=*/(short)0, acc[i][j],
            /*reuse_a=*/false, /*reuse_b=*/false);
      }
    }

    // drain the in-flight DMA (issuing waves) and publish; also ensures all
    // waves finished reading buf before the next iteration overwrites it
    __builtin_amdgcn_s_wait_tensorcnt(0);
    __syncthreads();
  }

  // -------- epilogue: C VGPR r -> M = r + 8*lhalf, N = lrow --------
#pragma unroll
  for (int i = 0; i < 2; ++i) {
#pragma unroll
    for (int j = 0; j < 4; ++j) {
      int col  = blockN + nw * 64 + j * 16 + lrow;
      int row0 = blockM + mw * 32 + i * 16 + lhalf * 8;
      if (col < N) {
#pragma unroll
        for (int r = 0; r < 8; ++r) {
          int row = row0 + r;
          float v = acc[i][j][r];
          if (mode == 1) {
            v += res[(size_t)row * N + col];
          } else if (mode == 2) {
            v += bias[col];
            v = (v > 20.0f) ? v : log1pf(expf(v));   // softplus
          }
          C[(size_t)row * N + col] = v;
        }
      }
    }
  }
}

// =====================================================================
// LayerNorm over D=1024, one block per token row; writes f16 for GEMM A.
// =====================================================================
__global__ __launch_bounds__(256) void ln_kernel(
    const float* __restrict__ x, const float* __restrict__ w,
    const float* __restrict__ b, _Float16* __restrict__ xn)
{
  const int row = blockIdx.x;
  const float* xr = x + (size_t)row * DMODEL;
  float s = 0.0f, sq = 0.0f;
  for (int i = threadIdx.x; i < DMODEL; i += 256) {
    float v = xr[i]; s += v; sq += v * v;
  }
#pragma unroll
  for (int m = 16; m >= 1; m >>= 1) {
    s  += __shfl_xor(s,  m, 32);
    sq += __shfl_xor(sq, m, 32);
  }
  __shared__ float ss[8], ssq[8];
  int wv = threadIdx.x >> 5, ln = threadIdx.x & 31;
  if (ln == 0) { ss[wv] = s; ssq[wv] = sq; }
  __syncthreads();
  s = 0.0f; sq = 0.0f;
#pragma unroll
  for (int i = 0; i < 8; ++i) { s += ss[i]; sq += ssq[i]; }
  float mean = s * (1.0f / DMODEL);
  float var  = sq * (1.0f / DMODEL) - mean * mean;
  float rstd = rsqrtf(var + 1e-5f);
  for (int i = threadIdx.x; i < DMODEL; i += 256) {
    float v = (xr[i] - mean) * rstd * w[i] + b[i];
    xn[(size_t)row * DMODEL + i] = (_Float16)v;
  }
}

// f32 -> f16 weight conversion
__global__ void cvt_f16_kernel(const float* __restrict__ src,
                               _Float16* __restrict__ dst, int n)
{
  int i = blockIdx.x * 256 + threadIdx.x;
  if (i < n) dst[i] = (_Float16)src[i];
}

// slice dt_low (cols 0..63 of x_dbl[ROWS,96]) to f16 GEMM A
__global__ void dtlow_kernel(const float* __restrict__ xdbl,
                             _Float16* __restrict__ dtlow)
{
  int i = blockIdx.x * 256 + threadIdx.x;   // ROWS*DTRANK
  int row = i / DTRANK, col = i % DTRANK;
  dtlow[i] = (_Float16)xdbl[(size_t)row * (DTRANK + 2 * DSTATE) + col];
}

// =====================================================================
// Depthwise causal conv1d (K=4) + SiLU.  Input = xs half of xz.
// One thread per (b,l,d).
// =====================================================================
__global__ __launch_bounds__(256) void conv_silu_kernel(
    const float* __restrict__ xz, const float* __restrict__ cw,
    const float* __restrict__ cb, float* __restrict__ xs_f,
    _Float16* __restrict__ xs_h)
{
  int idx = blockIdx.x * 256 + threadIdx.x;     // ROWS * DINNER
  int d  = idx & (DINNER - 1);
  int bl = idx >> 11;                           // token row (DINNER=2048)
  int l  = bl % LSEQ;
  float acc = cb[d];
#pragma unroll
  for (int k = 0; k < DCONV; ++k) {
    int ll = l + k - (DCONV - 1);
    if (ll >= 0)
      acc += cw[d * DCONV + k] * xz[(size_t)(bl + k - (DCONV - 1)) * (2 * DINNER) + d];
  }
  float v = acc / (1.0f + expf(-acc));          // SiLU
  xs_f[idx] = v;
  xs_h[idx] = (_Float16)v;
}

// =====================================================================
// Selective scan + D-skip + z-gate.  16 lanes per (b,d) channel, one per
// state; y contraction reduced with 4 shfl_xor hops within the 16-lane
// group.  Writes gated y as f16 (A-matrix of the out_proj GEMM).
// =====================================================================
__global__ __launch_bounds__(256) void scan_kernel(
    const float* __restrict__ dt, const float* __restrict__ xs_f,
    const float* __restrict__ xdbl, const float* __restrict__ xz,
    const float* __restrict__ A_log, const float* __restrict__ D_skip,
    _Float16* __restrict__ y_h)
{
  const int t  = threadIdx.x;
  const int s  = t & (DSTATE - 1);
  const int ch = blockIdx.x * 16 + (t >> 4);    // 0 .. BATCH*DINNER-1
  const int d  = ch & (DINNER - 1);
  const int b  = ch >> 11;
  const float Av = -expf(A_log[d * DSTATE + s]);
  const float Dv = D_skip[d];
  float h = 0.0f;
  const size_t base = (size_t)b * LSEQ;
  for (int l = 0; l < LSEQ; ++l) {
    size_t rl = base + l;
    float dtv = dt[rl * DINNER + d];
    float xv  = xs_f[rl * DINNER + d];
    float Bv  = xdbl[rl * (DTRANK + 2 * DSTATE) + DTRANK + s];
    float Cv  = xdbl[rl * (DTRANK + 2 * DSTATE) + DTRANK + DSTATE + s];
    float dA  = __expf(dtv * Av);
    h = dA * h + dtv * Bv * xv;
    float p = h * Cv;
    p += __shfl_xor(p, 1, 32);
    p += __shfl_xor(p, 2, 32);
    p += __shfl_xor(p, 4, 32);
    p += __shfl_xor(p, 8, 32);
    if (s == 0) {
      float y  = p + xv * Dv;
      float zv = xz[rl * (2 * DINNER) + DINNER + d];
      float g  = zv / (1.0f + expf(-zv));        // SiLU(z)
      y_h[rl * DINNER + d] = (_Float16)(y * g);
    }
  }
}

// =====================================================================
// Host-side orchestration
// =====================================================================
static inline size_t ws_take(size_t& off, size_t bytes) {
  size_t o = off;
  off = (off + bytes + 255) & ~(size_t)255;
  return o;
}

extern "C" void kernel_launch(void* const* d_in, const int* in_sizes, int n_in,
                              void* d_out, int out_size, void* d_ws, size_t ws_size,
                              hipStream_t stream) {
  (void)in_sizes; (void)n_in; (void)out_size; (void)ws_size;
  const float* x      = (const float*)d_in[0];
  const float* ln_w   = (const float*)d_in[1];
  const float* ln_b   = (const float*)d_in[2];
  const float* W_in   = (const float*)d_in[3];
  const float* conv_w = (const float*)d_in[4];
  const float* conv_b = (const float*)d_in[5];
  const float* W_xprj = (const float*)d_in[6];
  const float* W_dt   = (const float*)d_in[7];
  const float* b_dt   = (const float*)d_in[8];
  const float* A_log  = (const float*)d_in[9];
  const float* D_skip = (const float*)d_in[10];
  const float* W_out  = (const float*)d_in[11];
  float* out = (float*)d_out;

  char* ws = (char*)d_ws;
  size_t off = 0;
  _Float16* xn_h   = (_Float16*)(ws + ws_take(off, (size_t)ROWS * DMODEL * 2));
  _Float16* Win_h  = (_Float16*)(ws + ws_take(off, (size_t)(2 * DINNER) * DMODEL * 2));
  _Float16* Wxp_h  = (_Float16*)(ws + ws_take(off, (size_t)(DTRANK + 2 * DSTATE) * DINNER * 2));
  _Float16* Wdt_h  = (_Float16*)(ws + ws_take(off, (size_t)DINNER * DTRANK * 2));
  _Float16* Wout_h = (_Float16*)(ws + ws_take(off, (size_t)DMODEL * DINNER * 2));
  float*    xz     = (float*)   (ws + ws_take(off, (size_t)ROWS * 2 * DINNER * 4));
  float*    xs_f   = (float*)   (ws + ws_take(off, (size_t)ROWS * DINNER * 4));
  _Float16* xs_h   = (_Float16*)(ws + ws_take(off, (size_t)ROWS * DINNER * 2));
  float*    xdbl   = (float*)   (ws + ws_take(off, (size_t)ROWS * (DTRANK + 2 * DSTATE) * 4));
  _Float16* dtlo_h = (_Float16*)(ws + ws_take(off, (size_t)ROWS * DTRANK * 2));
  float*    dt_f   = (float*)   (ws + ws_take(off, (size_t)ROWS * DINNER * 4));
  _Float16* y_h    = (_Float16*)(ws + ws_take(off, (size_t)ROWS * DINNER * 2));

  // ---- weight conversions to f16 ----
  auto cvt = [&](const float* s, _Float16* d, int n) {
    cvt_f16_kernel<<<(n + 255) / 256, 256, 0, stream>>>(s, d, n);
  };
  cvt(W_in,   Win_h,  2 * DINNER * DMODEL);
  cvt(W_xprj, Wxp_h, (DTRANK + 2 * DSTATE) * DINNER);
  cvt(W_dt,   Wdt_h,  DINNER * DTRANK);
  cvt(W_out,  Wout_h, DMODEL * DINNER);

  // ---- LayerNorm ----
  ln_kernel<<<ROWS, 256, 0, stream>>>(x, ln_w, ln_b, xn_h);

  // ---- in_proj: xz[ROWS, 2*DINNER] ----
  wmma_gemm_kernel<<<dim3((2 * DINNER) / TILE_N, ROWS / TILE_M), 256, 0, stream>>>(
      xn_h, Win_h, xz, ROWS, 2 * DINNER, DMODEL, 0, nullptr, nullptr);

  // ---- depthwise conv + SiLU ----
  conv_silu_kernel<<<(ROWS * DINNER) / 256, 256, 0, stream>>>(
      xz, conv_w, conv_b, xs_f, xs_h);

  // ---- x_proj: xdbl[ROWS, 96] ----
  wmma_gemm_kernel<<<dim3((DTRANK + 2 * DSTATE + TILE_N - 1) / TILE_N, ROWS / TILE_M),
                    256, 0, stream>>>(
      xs_h, Wxp_h, xdbl, ROWS, DTRANK + 2 * DSTATE, DINNER, 0, nullptr, nullptr);

  // ---- dt_low slice -> f16 ----
  dtlow_kernel<<<(ROWS * DTRANK) / 256, 256, 0, stream>>>(xdbl, dtlo_h);

  // ---- dt_proj + softplus: dt_f[ROWS, DINNER] ----
  wmma_gemm_kernel<<<dim3(DINNER / TILE_N, ROWS / TILE_M), 256, 0, stream>>>(
      dtlo_h, Wdt_h, dt_f, ROWS, DINNER, DTRANK, 2, b_dt, nullptr);

  // ---- selective scan + D-skip + z-gate -> y_h f16 ----
  scan_kernel<<<(BATCH * DINNER) / 16, 256, 0, stream>>>(
      dt_f, xs_f, xdbl, xz, A_log, D_skip, y_h);

  // ---- out_proj + residual: out[ROWS, DMODEL] ----
  wmma_gemm_kernel<<<dim3(DMODEL / TILE_N, ROWS / TILE_M), 256, 0, stream>>>(
      y_h, Wout_h, out, ROWS, DMODEL, DINNER, 1, nullptr, x);
}
